// SpatialMix_91250875171039
// MI455X (gfx1250) — compile-verified, matrix operand order checked by reading, not run on hardware
//
#include <hip/hip_runtime.h>
#include <hip/hip_bf16.h>

// ---------------- shapes (specialized to the reference setup) ----------------
#define B_    8
#define H_    64
#define W_    64
#define T_    4096          // H_*W_
#define C_    512
#define M_    32768         // B_*T_
#define CH_   32            // scan chunk length
#define NC_   128           // T_/CH_
#define BC_   4096          // B_*C_
#define NEG_INF_ (-1.0e30f)

#define LDS_STRIDE 40       // halves per row: 80 B, 16B-aligned rows for async b128
#define NKT (C_ / 32)       // 16 K-tiles

typedef __attribute__((ext_vector_type(16))) __bf16 v16bf;
typedef __attribute__((ext_vector_type(8)))  float  v8f;
typedef __attribute__((ext_vector_type(8)))  int    v8i;

__device__ __forceinline__ unsigned short f2bf(float f) {
  unsigned int u = __builtin_bit_cast(unsigned int, f);
  u += 0x7FFFu + ((u >> 16) & 1u);          // round-to-nearest-even
  return (unsigned short)(u >> 16);
}

// ---- CDNA5 async global->LDS (ASYNCcnt-tracked, 16B/lane per instruction) ----
__device__ __forceinline__ void async_copy_b128(unsigned short* ldsDst,
                                                const unsigned short* g) {
  unsigned int off = (unsigned int)(unsigned long long)ldsDst; // LDS offset = low 32 bits
  asm volatile("global_load_async_to_lds_b128 %0, %1, off"
               :: "v"(off), "v"(g) : "memory");
}
template <int N>
__device__ __forceinline__ void wait_async() {
  asm volatile("s_wait_asynccnt %0" :: "i"(N) : "memory");
}

// ---------------- weight fp32 -> bf16 conversion (one-time, 2 MB) ----------------
__global__ __launch_bounds__(256) void convert_weights(
    const float* __restrict__ Wk, const float* __restrict__ Wv,
    const float* __restrict__ Wr, const float* __restrict__ Wo,
    unsigned short* __restrict__ dst) {
  int i = blockIdx.x * 256 + threadIdx.x;        // 4 * 512*512 elements
  int which = i >> 18;                           // 262144 per matrix
  int j = i & 262143;
  const float* src = (which == 0) ? Wk : (which == 1) ? Wv : (which == 2) ? Wr : Wo;
  dst[i] = f2bf(src[j]);
}

// ---------------- bf16 WMMA GEMM: out[m,n] = sum_c A[m,c] * W[n,c] ----------------
// Macro tile 128(M) x 256(N); 8 waves, 64x64 per wave -> 16 WMMAs/K-step.
// FUSED: A = mix[c]*x + (1-mix[c])*qshift(x) built on the fly (reg-prefetched).
// !FUSED: A bf16 staged via async global->LDS.  B always async.  Double buffered.
template <int FUSED, int SIG>
__global__ __launch_bounds__(256) void gemm_wmma(
    const float* __restrict__ x,
    const unsigned short* __restrict__ Abf,
    const unsigned short* __restrict__ Wbf,      // bf16 weights [N=512][K=512]
    const float* __restrict__ mix,
    float* __restrict__ outF)
{
  __shared__ unsigned short ldsA[2][128 * LDS_STRIDE];   // 2 x 10 KB
  __shared__ unsigned short ldsB[2][256 * LDS_STRIDE];   // 2 x 20 KB

  const int tid  = threadIdx.x;
  const int lane = tid & 31;
  const int wid  = tid >> 5;
  const int l16  = lane & 15;
  const int half = lane >> 4;
  const int waveM = wid & 1;                     // 2 x 64-row halves
  const int waveN = wid >> 1;                    // 4 x 64-col strips
  const int m0 = blockIdx.x * 128;
  const int n0 = blockIdx.y * 256;

  const int rA  = tid >> 1;                      // 0..127: A-tile row this thread stages
  const int cHf = tid & 1;                       // 16-col half

  // per-thread invariants for the fused q_shift gather
  const int m = m0 + rA;
  const int b = m >> 12;
  const int t = m & (T_ - 1);
  const int h = t >> 6;
  const int w = t & (W_ - 1);

  v8f acc[4][4] = {};

  float4 pxa[4], pxs[4], pmx[4];                 // fused-mode prefetch registers

  // per-tile async instruction count (per wave): B rows 4x b128 [+2 for plain A]
  constexpr int NASYNC = FUSED ? 4 : 6;

  // ---- stage issue: start moving K-tile kt into LDS buffer `buf` ----
  auto stage_issue = [&](int kt, int buf) {
    const int kk0 = kt * 32;
    // B tile: ldsB[buf][n][k] = W[n0+n][kk0+k]; one 32-col row per thread (4x b128)
    {
      const unsigned short* src = Wbf + (long)(n0 + tid) * C_ + kk0;
      unsigned short* dst = &ldsB[buf][tid * LDS_STRIDE];
      async_copy_b128(dst, src);
      async_copy_b128(dst + 8, src + 8);
      async_copy_b128(dst + 16, src + 16);
      async_copy_b128(dst + 24, src + 24);
    }
    if (FUSED) {
      const int g  = kk0 >> 7;                   // uniform q_shift group
      const int dw = (g == 0) ? -1 : (g == 1) ? 1 : 0;
      const int dh = (g == 2) ? -1 : (g == 3) ? 1 : 0;
      const int wn = w + dw, hn = h + dh;
      const bool valid = ((unsigned)wn < (unsigned)W_) && ((unsigned)hn < (unsigned)H_);
      const long msrc = ((long)b << 12) + ((long)(valid ? hn : h) << 6) + (valid ? wn : w);
      const float* xr = x + (long)m * C_ + kk0 + cHf * 16;
      const float* xs = x + msrc * C_ + kk0 + cHf * 16;
      const float* mp = mix + kk0 + cHf * 16;
      #pragma unroll
      for (int i = 0; i < 4; ++i) {
        pxa[i] = *(const float4*)(xr + 4 * i);
        float4 s4 = *(const float4*)(xs + 4 * i);
        if (!valid) s4 = make_float4(0.f, 0.f, 0.f, 0.f);
        pxs[i] = s4;
        pmx[i] = *(const float4*)(mp + 4 * i);
      }
    } else {
      const unsigned short* src = Abf + (long)m * C_ + kk0 + cHf * 16;
      unsigned short* dst = &ldsA[buf][rA * LDS_STRIDE + cHf * 16];
      async_copy_b128(dst, src);
      async_copy_b128(dst + 8, src + 8);
    }
  };

  // ---- stage commit (fused only): convert prefetched regs, store to LDS ----
  auto stage_commit = [&](int buf) {
    if (FUSED) {
      unsigned short* dst = &ldsA[buf][rA * LDS_STRIDE + cHf * 16];
      #pragma unroll
      for (int i = 0; i < 4; ++i) {
        dst[4 * i + 0] = f2bf(pxa[i].x * pmx[i].x + pxs[i].x * (1.f - pmx[i].x));
        dst[4 * i + 1] = f2bf(pxa[i].y * pmx[i].y + pxs[i].y * (1.f - pmx[i].y));
        dst[4 * i + 2] = f2bf(pxa[i].z * pmx[i].z + pxs[i].z * (1.f - pmx[i].z));
        dst[4 * i + 3] = f2bf(pxa[i].w * pmx[i].w + pxs[i].w * (1.f - pmx[i].w));
      }
    }
  };

  // prologue: stage tile 0 into buffer 0
  stage_issue(0, 0);
  stage_commit(0);

  for (int kt = 0; kt < NKT; ++kt) {
    const int buf = kt & 1;
    // issue tile kt+1 first; in-order async completion lets us wait for only
    // the older tile (ASYNCcnt <= NASYNC) while kt+1's transfers keep flying.
    if (kt + 1 < NKT) {
      stage_issue(kt + 1, buf ^ 1);
      wait_async<NASYNC>();
    } else {
      wait_async<0>();
    }
    __syncthreads();               // all waves' tile-kt writes visible

    // ---- fragments (ISA 7.12.2 16-bit A layout) + 16 WMMAs ----
    v16bf afr[4], bfr[4];
    #pragma unroll
    for (int ms = 0; ms < 4; ++ms) {
      const int base = (waveM * 64 + ms * 16 + l16) * LDS_STRIDE;
      v8i tv;
      #pragma unroll
      for (int j = 0; j < 8; ++j) {
        const int k0 = (j < 4) ? (2 * j + 8 * half) : (16 + 2 * (j - 4) + 8 * half);
        tv[j] = *(const int*)&ldsA[buf][base + k0];
      }
      afr[ms] = __builtin_bit_cast(v16bf, tv);
    }
    #pragma unroll
    for (int ns = 0; ns < 4; ++ns) {
      const int base = (waveN * 64 + ns * 16 + l16) * LDS_STRIDE;
      v8i tv;
      #pragma unroll
      for (int j = 0; j < 8; ++j) {
        const int k0 = (j < 4) ? (2 * j + 8 * half) : (16 + 2 * (j - 4) + 8 * half);
        tv[j] = *(const int*)&ldsB[buf][base + k0];
      }
      bfr[ns] = __builtin_bit_cast(v16bf, tv);
    }
    #pragma unroll
    for (int ms = 0; ms < 4; ++ms)
      #pragma unroll
      for (int ns = 0; ns < 4; ++ns)
        acc[ms][ns] = __builtin_amdgcn_wmma_f32_16x16x32_bf16(
            false, afr[ms], false, bfr[ns], (short)0, acc[ms][ns], false, false);

    if (kt + 1 < NKT) stage_commit(buf ^ 1);
  }

  // ---- epilogue: C/D layout -> VGPR r holds M = r + 8*half, N = lane&15 ----
  #pragma unroll
  for (int ms = 0; ms < 4; ++ms) {
    const int rowBase = m0 + waveM * 64 + ms * 16 + half * 8;
    #pragma unroll
    for (int ns = 0; ns < 4; ++ns) {
      const int col = n0 + waveN * 64 + ns * 16 + l16;
      #pragma unroll
      for (int r = 0; r < 8; ++r) {
        float vv = acc[ms][ns][r];
        if (SIG) vv = 1.f / (1.f + __expf(-vv));
        outF[(long)(rowBase + r) * C_ + col] = vv;
      }
    }
  }
}

// ---------------- phase 1: per-chunk (fwd,bwd) WKV summaries ----------------
__global__ __launch_bounds__(128) void wkv_chunk_summaries(
    const float* __restrict__ k, const float* __restrict__ v,
    const float* __restrict__ decay,
    float* __restrict__ fsp, float* __restrict__ fsa, float* __restrict__ fsb,
    float* __restrict__ bsp, float* __restrict__ bsa, float* __restrict__ bsb)
{
  const int c  = blockIdx.x * 128 + threadIdx.x;
  const int ch = blockIdx.y;
  const int b  = blockIdx.z;
  const float w = decay[c] * (1.0f / T_);
  const long base = ((long)b * T_ + (long)ch * CH_) * C_ + c;
  const int idx = ch * BC_ + b * C_ + c;

  float p = NEG_INF_, a = 0.f, bb = 0.f;
  for (int i = 0; i < CH_; ++i) {
    float kt = k[base + (long)i * C_];
    float vt = v[base + (long)i * C_];
    float p2 = p - w;
    float q  = fmaxf(p2, kt);
    float e1 = __expf(p2 - q), e2 = __expf(kt - q);
    a = a * e1 + e2 * vt; bb = bb * e1 + e2; p = q;
  }
  fsp[idx] = p; fsa[idx] = a; fsb[idx] = bb;

  p = NEG_INF_; a = 0.f; bb = 0.f;
  for (int i = CH_ - 1; i >= 0; --i) {
    float kt = k[base + (long)i * C_];
    float vt = v[base + (long)i * C_];
    float p2 = p - w;
    float q  = fmaxf(p2, kt);
    float e1 = __expf(p2 - q), e2 = __expf(kt - q);
    a = a * e1 + e2 * vt; bb = bb * e1 + e2; p = q;
  }
  bsp[idx] = p; bsa[idx] = a; bsb[idx] = bb;
}

// ---------------- phase 2: exclusive scan over chunk summaries ----------------
__device__ __forceinline__ void compose(float& P, float& A, float& Bv,
                                        float ps, float as, float bs, float Lw) {
  float pd = P - Lw;
  float q  = fmaxf(pd, ps);
  float e1 = __expf(pd - q), e2 = __expf(ps - q);
  A = A * e1 + as * e2;
  Bv = Bv * e1 + bs * e2;
  P = q;
}

__global__ __launch_bounds__(256) void wkv_scan_chunks(
    const float* __restrict__ decay,
    const float* __restrict__ fsp, const float* __restrict__ fsa, const float* __restrict__ fsb,
    const float* __restrict__ bsp, const float* __restrict__ bsa, const float* __restrict__ bsb,
    float* __restrict__ efp, float* __restrict__ efa, float* __restrict__ efb,
    float* __restrict__ ebp, float* __restrict__ eba, float* __restrict__ ebb)
{
  const int bc = blockIdx.x * 256 + threadIdx.x;   // b*C + c
  const int c = bc & (C_ - 1);
  const float Lw = (decay[c] * (1.0f / T_)) * CH_;

  float P = NEG_INF_, A = 0.f, Bv = 0.f;
  for (int ch = 0; ch < NC_; ++ch) {
    const int idx = ch * BC_ + bc;
    efp[idx] = P; efa[idx] = A; efb[idx] = Bv;
    compose(P, A, Bv, fsp[idx], fsa[idx], fsb[idx], Lw);
  }
  P = NEG_INF_; A = 0.f; Bv = 0.f;
  for (int ch = NC_ - 1; ch >= 0; --ch) {
    const int idx = ch * BC_ + bc;
    ebp[idx] = P; eba[idx] = A; ebb[idx] = Bv;
    compose(P, A, Bv, bsp[idx], bsa[idx], bsb[idx], Lw);
  }
}

// ---------------- phase 3: replay chunk, combine fwd/bwd/bonus, apply sigmoid(r) ----------------
__global__ __launch_bounds__(128) void wkv_finalize(
    const float* __restrict__ k, const float* __restrict__ v,
    const float* __restrict__ sr,
    const float* __restrict__ decay, const float* __restrict__ first,
    const float* __restrict__ efp, const float* __restrict__ efa, const float* __restrict__ efb,
    const float* __restrict__ ebp, const float* __restrict__ eba, const float* __restrict__ ebb,
    unsigned short* __restrict__ sy)
{
  __shared__ float lds[CH_ * 3 * 128];             // backward states, 48 KB
  const int tid = threadIdx.x;
  const int c  = blockIdx.x * 128 + tid;
  const int ch = blockIdx.y;
  const int b  = blockIdx.z;
  const float w = decay[c] * (1.0f / T_);
  const float u = first[c] * (1.0f / T_);
  const int idx = ch * BC_ + b * C_ + c;
  const long base = ((long)b * T_ + (long)ch * CH_) * C_ + c;

  // right-to-left: stash exclusive backward states for this chunk in LDS
  float p = ebp[idx], a = eba[idx], bb = ebb[idx];
  for (int i = CH_ - 1; i >= 0; --i) {
    lds[(i * 3 + 0) * 128 + tid] = p;
    lds[(i * 3 + 1) * 128 + tid] = a;
    lds[(i * 3 + 2) * 128 + tid] = bb;
    float kt = k[base + (long)i * C_];
    float vt = v[base + (long)i * C_];
    float p2 = p - w;
    float q  = fmaxf(p2, kt);
    float e1 = __expf(p2 - q), e2 = __expf(kt - q);
    a = a * e1 + e2 * vt; bb = bb * e1 + e2; p = q;
  }

  // left-to-right: forward replay + combine
  p = efp[idx]; a = efa[idx]; bb = efb[idx];
  for (int i = 0; i < CH_; ++i) {
    float kt = k[base + (long)i * C_];
    float vt = v[base + (long)i * C_];
    float pr = lds[(i * 3 + 0) * 128 + tid];
    float ar = lds[(i * 3 + 1) * 128 + tid];
    float br = lds[(i * 3 + 2) * 128 + tid];
    float pc = u + kt;
    float q  = fmaxf(fmaxf(p, pr), pc);
    float el = __expf(p - q), er = __expf(pr - q), ec = __expf(pc - q);
    float num = a * el + ar * er + ec * vt;
    float den = bb * el + br * er + ec;
    float y = num / den;
    sy[base + (long)i * C_] = f2bf(sr[base + (long)i * C_] * y);
    // advance forward state
    float p2 = p - w;
    float q2 = fmaxf(p2, kt);
    float e1 = __expf(p2 - q2), e2 = __expf(kt - q2);
    a = a * e1 + e2 * vt; bb = bb * e1 + e2; p = q2;
  }
}

// ---------------- host orchestration ----------------
extern "C" void kernel_launch(void* const* d_in, const int* in_sizes, int n_in,
                              void* d_out, int out_size, void* d_ws, size_t ws_size,
                              hipStream_t stream) {
  const float* x     = (const float*)d_in[0];
  const float* mix_k = (const float*)d_in[3];
  const float* mix_v = (const float*)d_in[4];
  const float* mix_r = (const float*)d_in[5];
  const float* Wk    = (const float*)d_in[6];
  const float* Wv    = (const float*)d_in[7];
  const float* Wr    = (const float*)d_in[8];
  const float* Wo    = (const float*)d_in[9];
  const float* decay = (const float*)d_in[10];
  const float* first = (const float*)d_in[11];
  float* out = (float*)d_out;

  char* p = (char*)d_ws;
  auto take = [&](size_t bytes) {
    char* r = p;
    p += (bytes + 255) & ~(size_t)255;
    return r;
  };

  unsigned short* wbf = (unsigned short*)take((size_t)4 * 512 * 512 * 2);   // 2 MB
  unsigned short* wkb = wbf;
  unsigned short* wvb = wbf + 1 * 262144;
  unsigned short* wrb = wbf + 2 * 262144;
  unsigned short* wob = wbf + 3 * 262144;
  float* kf  = (float*)take((size_t)M_ * C_ * 4);                            // 64 MB
  float* vf  = (float*)take((size_t)M_ * C_ * 4);                            // 64 MB
  float* srf = (float*)take((size_t)M_ * C_ * 4);                            // 64 MB
  unsigned short* syb = (unsigned short*)take((size_t)M_ * C_ * 2);          // 32 MB
  float* fsp = (float*)take((size_t)NC_ * BC_ * 4);
  float* fsa = (float*)take((size_t)NC_ * BC_ * 4);
  float* fsb = (float*)take((size_t)NC_ * BC_ * 4);
  float* bsp = (float*)take((size_t)NC_ * BC_ * 4);
  float* bsa = (float*)take((size_t)NC_ * BC_ * 4);
  float* bsb = (float*)take((size_t)NC_ * BC_ * 4);
  float* efp = (float*)take((size_t)NC_ * BC_ * 4);
  float* efa = (float*)take((size_t)NC_ * BC_ * 4);
  float* efb = (float*)take((size_t)NC_ * BC_ * 4);
  float* ebp = (float*)take((size_t)NC_ * BC_ * 4);
  float* eba = (float*)take((size_t)NC_ * BC_ * 4);
  float* ebb = (float*)take((size_t)NC_ * BC_ * 4);

  convert_weights<<<4096, 256, 0, stream>>>(Wk, Wv, Wr, Wo, wbf);

  dim3 gg(M_ / 128, C_ / 256);   // (256, 2)
  gemm_wmma<1, 0><<<gg, 256, 0, stream>>>(x, nullptr, wkb, mix_k, kf);
  gemm_wmma<1, 0><<<gg, 256, 0, stream>>>(x, nullptr, wvb, mix_v, vf);
  gemm_wmma<1, 1><<<gg, 256, 0, stream>>>(x, nullptr, wrb, mix_r, srf);

  wkv_chunk_summaries<<<dim3(C_ / 128, NC_, B_), 128, 0, stream>>>(
      kf, vf, decay, fsp, fsa, fsb, bsp, bsa, bsb);
  wkv_scan_chunks<<<BC_ / 256, 256, 0, stream>>>(
      decay, fsp, fsa, fsb, bsp, bsa, bsb, efp, efa, efb, ebp, eba, ebb);
  wkv_finalize<<<dim3(C_ / 128, NC_, B_), 128, 0, stream>>>(
      kf, vf, srf, decay, first, efp, efa, efb, ebp, eba, ebb, syb);

  gemm_wmma<0, 0><<<gg, 256, 0, stream>>>(nullptr, syb, wob, nullptr, out);
}
